// SVDRnn_3624952398509
// MI455X (gfx1250) — matrix operand order repaired and checked by reading, not compiled, not run on hardware
//
#include <hip/hip_runtime.h>
#include <hip/hip_bf16.h>

// ---------------------------------------------------------------------------
// SVD-RNN for MI455X (gfx1250, wave32, WMMA bf16, TDM weight staging)
//   B=64, T=2048, I=128, H=256, O=128, K1=K2=1, M=256
// Weights are stored PRE-SWIZZLED in WMMA B-fragment order:
//   fragment (32x16 KxN tile) = 512 bf16 = 1024 B, lane-major, 16 bf16/lane:
//     lane<16 : N=n0+lane, K=k0+(e&7)+16*(e>>3)   (e=0..15)
//     lane>=16: N=n0+lane-16, K=k0+8+(e&7)+16*(e>>3)
// so each lane loads its whole fragment as one aligned 32-byte LDS access.
// W_outT fragments (loop-invariant, 8 per wave) are hoisted into registers.
// ---------------------------------------------------------------------------

#define NB   64
#define NT   2048
#define NI   128
#define NH   256
#define NO   128
#define RB   16          // batch rows per workgroup in the recurrence kernel

typedef __bf16 bf16_t;
typedef __bf16 v16bf __attribute__((ext_vector_type(16)));
typedef __bf16 v8bf  __attribute__((ext_vector_type(8)));
typedef float  v8f   __attribute__((ext_vector_type(8)));
typedef unsigned int u32x4 __attribute__((ext_vector_type(4)));
typedef int          i32x8 __attribute__((ext_vector_type(8)));
typedef int          i32x4 __attribute__((ext_vector_type(4)));

// ---- workspace layout (bytes) ----
#define WS_U      0                // 256*256 f32 = 256 KB
#define WS_V      262144           // 256*256 f32 = 256 KB
#define WS_WT     524288           // 65536 bf16 (128 frags: 8 kc x 16 nt)
#define WS_WINT   655360           // 32768 bf16 ( 64 frags: 4 kc x 16 nt)
#define WS_WOUTT  720896           // 32768 bf16 ( 64 frags: 8 kc x  8 nt)
#define WS_TOTAL  786432

// ---- recurrence-kernel LDS layout (dynamic) ----
#define L_WT      0                // 131072
#define L_WINT    131072           // 65536
#define L_WOUTT   196608           // 65536
#define L_H       262144           // 16*256*2 = 8192
#define L_X       270336           // 2 * 16*128*2 = 8192
#define L_BIN     278528           // 1024
#define L_BOUT    279552           // 512
#define L_RNN_BYTES 280064

#define L_UV_BYTES (262144 + 1024 + 1024)

// ===========================================================================
// WMMA helpers (layouts per CDNA5 ISA 7.12.2, wave32)
// ===========================================================================
__device__ inline v8f wmma_bf16(v16bf a, v16bf b, v8f c) {
  return __builtin_amdgcn_wmma_f32_16x16x32_bf16(
      /*neg_a=*/false, a, /*neg_b=*/false, b,
      /*c_mod=*/(short)0, c, /*reuse_a=*/false, /*reuse_b=*/false);
}

__device__ inline v8f splat8(float v) {
  v8f r;
#pragma unroll
  for (int e = 0; e < 8; ++e) r[e] = v;
  return r;
}

// A-matrix 16x32 bf16 fragment from a row-major LDS tile: two aligned
// 16-byte loads per lane (-> ds_load_b128), concatenated in registers.
// lane<16 : M=lane,    K = {k0..k0+7, k0+16..k0+23}
// lane>=16: M=lane-16, K = {k0+8..k0+15, k0+24..k0+31}
__device__ inline v16bf ldsA_frag(const bf16_t* base, int stride, int k0) {
  int lane = threadIdx.x & 31;
  int r    = lane & 15;
  int hi   = (lane >> 4) & 1;
  const bf16_t* p = base + r * stride + k0 + hi * 8;
  v8bf lo = *(const v8bf*)p;
  v8bf hh = *(const v8bf*)(p + 16);
  return __builtin_shufflevector(lo, hh, 0, 1, 2, 3, 4, 5, 6, 7,
                                 8, 9, 10, 11, 12, 13, 14, 15);
}

// B-matrix fragment: pre-swizzled buffer, fragIdx in fragment-grid order.
// One aligned 32-byte access per lane (-> 2x ds_load_b128, no packing ALU).
__device__ inline v16bf ldsB_frag(const bf16_t* buf, int fragIdx) {
  return ((const v16bf*)buf)[fragIdx * 32 + (threadIdx.x & 31)];
}

// Decode a linear index inside a fragment-order buffer into (k, n).
__device__ inline void frag_decode(int idx, int ntiles, int& k, int& n) {
  int f     = idx >> 9;          // 512 elems per fragment
  int inner = idx & 511;
  int lane  = inner >> 4;
  int e     = inner & 15;
  int kc    = f / ntiles;
  int nt    = f % ntiles;
  k = kc * 32 + ((lane >> 4) & 1) * 8 + (e & 7) + (e >> 3) * 16;
  n = nt * 16 + (lane & 15);
}

// ===========================================================================
// TDM: 2D tensor load Global -> LDS (descriptor per CDNA5 ISA ch. 8)
// 6-arg builtin form: (u32x4 g0, i32x8 g1, i32x4, i32x4, i32x8, i32 cpol)
// ===========================================================================
#if __has_builtin(__builtin_amdgcn_tensor_load_to_lds)
#define HAVE_TDM 1
__device__ inline void tdm_load_2d(unsigned lds_off, const void* gptr,
                                   unsigned dim0, unsigned dim1,
                                   unsigned dsz_code /*0=1B,1=2B,2=4B*/) {
  unsigned long long ga = (unsigned long long)gptr;
  u32x4 g0;
  g0[0] = 1u;                                   // count=1, user descriptor
  g0[1] = lds_off;                              // lds_addr [63:32]
  g0[2] = (unsigned)ga;                         // global_addr low
  g0[3] = (unsigned)((ga >> 32) & 0x01FFFFFFu)  // global_addr [56:32]
        | (2u << 30);                           // type = 2 ("image")
  i32x8 g1;
  g1[0] = (int)(dsz_code << 16);                                   // mask=0, data_size
  g1[1] = (int)((dim0 & 0xFFFFu) << 16);                           // tensor_dim0 lo16
  g1[2] = (int)(((dim0 >> 16) & 0xFFFFu) | ((dim1 & 0xFFFFu) << 16)); // d0 hi | d1 lo
  g1[3] = (int)(((dim1 >> 16) & 0xFFFFu) | ((dim0 & 0xFFFFu) << 16)); // d1 hi | tile_dim0
  g1[4] = (int)(dim1 & 0xFFFFu);                                   // tile_dim1, tile_dim2=0
  g1[5] = (int)dim0;                                               // tensor_dim0_stride lo32
  g1[6] = 0;
  g1[7] = 0;
  i32x4 z4 = {0, 0, 0, 0};
  i32x8 z8 = {0, 0, 0, 0, 0, 0, 0, 0};
  __builtin_amdgcn_tensor_load_to_lds(g0, g1, z4, z4, z8, 0);
}
#else
#define HAVE_TDM 0
#endif

// ===========================================================================
// Kernel 1: Householder chains. block 0 -> U, block 1 -> V (run concurrently)
// ===========================================================================
__global__ __launch_bounds__(256, 1)
void build_uv_kernel(const float* __restrict__ u_raw,
                     const float* __restrict__ v_raw,
                     float* __restrict__ Uout, float* __restrict__ Vout) {
  extern __shared__ char smem[];
  float* Mm  = (float*)smem;                  // 256x256
  float* uv  = (float*)(smem + 262144);       // 256
  float* red = (float*)(smem + 263168);       // 256
  const int j = threadIdx.x;
  const bool isU = (blockIdx.x == 0);
  const float* raw = isU ? u_raw : v_raw;

  for (int i = 0; i < NH; ++i) Mm[i * NH + j] = (i == j) ? 1.0f : 0.0f;
  __syncthreads();

  for (int m = 0; m < NH; ++m) {
    float uj;
    if (isU) {                       // u_hats[m][j] = masked(u_raw[m]) flipped in cols
      int c = NH - 1 - j;
      uj = (c < 1 + m) ? raw[m * NH + c] : 0.0f;
    } else {                         // v_hats flipped in rows AND cols
      int r = NH - 1 - m, c = NH - 1 - j;
      uj = (c < 1 + r) ? raw[r * NH + c] : 0.0f;
    }
    uv[j]  = uj;
    red[j] = uj * uj;
    __syncthreads();
    for (int s = 128; s > 0; s >>= 1) {       // dot(u,u) tree reduction
      if (j < s) red[j] += red[j + s];
      __syncthreads();
    }
    float beta = 2.0f / red[0];
    // column j is private to thread j: t_j = (u @ M)_j, then M[:,j] -= beta*u*t_j
    float tj = 0.0f;
    for (int i = 0; i < NH; ++i) tj += uv[i] * Mm[i * NH + j];
    tj *= beta;
    for (int i = 0; i < NH; ++i) Mm[i * NH + j] -= uv[i] * tj;
    __syncthreads();
  }
  float* dst = isU ? Uout : Vout;
  for (int i = 0; i < NH; ++i) dst[i * NH + j] = Mm[i * NH + j];
}

// ===========================================================================
// Kernel 2: WT fragment buffer: element (k,n) = W[n][k] = sum_h U[n][h]*s[h]*V[h][k]
// ===========================================================================
__global__ __launch_bounds__(256)
void combine_w_kernel(const float* __restrict__ U, const float* __restrict__ V,
                      const float* __restrict__ sig, bf16_t* __restrict__ WT) {
  int idx = blockIdx.x * 256 + threadIdx.x;   // 0..65535
  int k, n;
  frag_decode(idx, 16, k, n);                 // 8 kc x 16 nt
  float acc = 0.0f;
  for (int h = 0; h < NH; ++h) acc += U[n * NH + h] * sig[h] * V[h * NH + k];
  WT[idx] = (bf16_t)acc;
}

// ===========================================================================
// Kernel 3: W_inT / W_outT fragment buffers (bf16)
// ===========================================================================
__global__ __launch_bounds__(256)
void transpose_aux_kernel(const float* __restrict__ W_in,
                          const float* __restrict__ W_out,
                          bf16_t* __restrict__ WinT, bf16_t* __restrict__ WoutT) {
  int e = blockIdx.x * 256 + threadIdx.x;     // 0..65535
  if (e < NI * NH) {                          // WinT: K=i(128), N=h(256): 4 kc x 16 nt
    int k, n;
    frag_decode(e, 16, k, n);
    WinT[e] = (bf16_t)W_in[n * NI + k];       // W_in[h][i]
  } else {                                    // WoutT: K=h(256), N=o(128): 8 kc x 8 nt
    int e2 = e - NI * NH;
    int k, n;
    frag_decode(e2, 8, k, n);
    WoutT[e2] = (bf16_t)W_out[n * NH + k];    // W_out[o][h]
  }
}

// ===========================================================================
// Kernel 4: fused recurrence. 4 WGs x 16 batch rows, all weights in LDS.
// Per step/wave: 8 WMMA (xproj) + 16 WMMA (h·Wᵀ) + 8 WMMA (output).
// ===========================================================================
// Each thread stages 8 contiguous x floats: 2x global_load_b128 -> cvt ->
// 1x ds_store_b128.
__device__ inline void stage_x(const float* __restrict__ x, bf16_t* xd,
                               int b0, int t) {
  int tid = threadIdx.x;
  int rr  = tid >> 4;                         // 0..15 (batch row)
  int c8  = (tid & 15) * 8;                   // column group of 8
  const float4* src =
      (const float4*)(x + (((long long)(b0 + rr)) * NT + t) * NI + c8);
  float4 f0 = src[0];
  float4 f1 = src[1];
  v8bf v;
  v[0] = (bf16_t)f0.x; v[1] = (bf16_t)f0.y;
  v[2] = (bf16_t)f0.z; v[3] = (bf16_t)f0.w;
  v[4] = (bf16_t)f1.x; v[5] = (bf16_t)f1.y;
  v[6] = (bf16_t)f1.z; v[7] = (bf16_t)f1.w;
  *(v8bf*)(xd + rr * NI + c8) = v;
}

__global__ __launch_bounds__(256, 1)
void rnn_kernel(const float* __restrict__ x,
                const bf16_t* __restrict__ gWT,
                const bf16_t* __restrict__ gWinT,
                const bf16_t* __restrict__ gWoutT,
                const float* __restrict__ b_in,
                const float* __restrict__ b_out,
                float* __restrict__ out) {
  extern __shared__ char smem[];
  bf16_t* sWT    = (bf16_t*)(smem + L_WT);
  bf16_t* sWinT  = (bf16_t*)(smem + L_WINT);
  bf16_t* sWoutT = (bf16_t*)(smem + L_WOUTT);
  bf16_t* s_h    = (bf16_t*)(smem + L_H);
  bf16_t* s_x    = (bf16_t*)(smem + L_X);
  float*  s_bin  = (float*)(smem + L_BIN);
  float*  s_bout = (float*)(smem + L_BOUT);

  const int tid  = threadIdx.x;
  const int wave = tid >> 5;
  const int lane = tid & 31;
  const int col  = lane & 15;
  const int hi   = (lane >> 4) & 1;
  const int b0   = blockIdx.x * RB;

  // ---- stage all weights into LDS once (TDM async DMA if available) ----
#if HAVE_TDM
  if (wave == 0) {
    tdm_load_2d((unsigned)(unsigned long long)(void*)sWT,    gWT,    512, 128, 1u);
    tdm_load_2d((unsigned)(unsigned long long)(void*)sWinT,  gWinT,  512, 64, 1u);
    tdm_load_2d((unsigned)(unsigned long long)(void*)sWoutT, gWoutT, 512, 64, 1u);
    __builtin_amdgcn_s_wait_tensorcnt(0);
  }
#else
  {
    const unsigned* s0 = (const unsigned*)gWT;     // 32768 dwords
    unsigned* d0 = (unsigned*)sWT;
    for (int i = tid; i < 32768; i += 256) d0[i] = s0[i];
    const unsigned* s1 = (const unsigned*)gWinT;   // 16384 dwords
    unsigned* d1 = (unsigned*)sWinT;
    for (int i = tid; i < 16384; i += 256) d1[i] = s1[i];
    const unsigned* s2 = (const unsigned*)gWoutT;  // 16384 dwords
    unsigned* d2 = (unsigned*)sWoutT;
    for (int i = tid; i < 16384; i += 256) d2[i] = s2[i];
  }
#endif
  s_bin[tid] = b_in[tid];
  if (tid < NO) s_bout[tid] = b_out[tid];
  stage_x(x, s_x, b0, 0);                      // x tile for t=0 into buffer 0
  __syncthreads();

  // h0 = 0: keep the 8 K-chunk A-fragments of h in registers
  v16bf aFr[8];
#pragma unroll
  for (int i = 0; i < 8; ++i) aFr[i] = (v16bf){};

  const int nt0 = wave * 2;                    // two N-tiles of H per wave
  const int nt1 = nt0 + 1;
  const int n0  = nt0 * 16;
  const int n1  = nt1 * 16;
  const int nto = wave;                        // one N-tile of O per wave
  const int no  = nto * 16;

  // loop-invariant biases, pre-splat into WMMA C initializers
  const float bin0 = s_bin[n0 + col];
  const float bin1 = s_bin[n1 + col];
  const float bo   = s_bout[no + col];
  const v8f cX0 = splat8(bin0);
  const v8f cX1 = splat8(bin1);
  const v8f cO  = splat8(bo);

  // hoist the 8 loop-invariant W_outT B-fragments into registers (64 VGPRs):
  // the post-barrier output GEMM then runs with zero LDS traffic.
  v16bf bO[8];
#pragma unroll
  for (int kc = 0; kc < 8; ++kc) bO[kc] = ldsB_frag(sWoutT, kc * 8 + nto);

  int buf = 0;

#pragma unroll 1
  for (int t = 0; t < NT; ++t) {
    const bf16_t* xA = s_x + buf * (RB * NI);

    // xproj chain (independent of h/barrier): accX = b_in + x_t @ W_inT
    v8f accX0 = cX0, accX1 = cX1;
#pragma unroll
    for (int kc = 0; kc < 4; ++kc) {
      v16bf a = ldsA_frag(xA, NI, kc * 32);
      accX0 = wmma_bf16(a, ldsB_frag(sWinT, kc * 16 + nt0), accX0);
      accX1 = wmma_bf16(a, ldsB_frag(sWinT, kc * 16 + nt1), accX1);
    }
    // recurrent chain: accR = h @ WT   (critical path from aFr)
    v8f accR0 = {}, accR1 = {};
#pragma unroll
    for (int kc = 0; kc < 8; ++kc) {
      accR0 = wmma_bf16(aFr[kc], ldsB_frag(sWT, kc * 16 + nt0), accR0);
      accR1 = wmma_bf16(aFr[kc], ldsB_frag(sWT, kc * 16 + nt1), accR1);
    }

    // h_new = tanh(accX + accR) -> LDS (C layout: elem r -> row r+8*hi, col)
#pragma unroll
    for (int r = 0; r < 8; ++r) {
      int row = r + hi * 8;
      s_h[row * NH + n0 + col] = (bf16_t)tanhf(accX0[r] + accR0[r]);
      s_h[row * NH + n1 + col] = (bf16_t)tanhf(accX1[r] + accR1[r]);
    }

    // double-buffered stage of x for t+1; prefetch t+2
    if (t + 1 < NT) {
      stage_x(x, s_x + (buf ^ 1) * (RB * NI), b0, t + 1);
      if (t + 2 < NT) {
        const float* pf =
            &x[(((long long)(b0 + (tid >> 4))) * NT + (t + 2)) * NI];
        __builtin_prefetch(pf, 0, 1);
      }
    }
    __syncthreads();

    // reload h_{t+1} A-fragments (also feeds next step's recurrent GEMM)
#pragma unroll
    for (int kc = 0; kc < 8; ++kc) aFr[kc] = ldsA_frag(s_h, NH, kc * 32);
    __syncthreads();

    // out_t = b_out + h_{t+1} @ W_outT : pure register WMMA chain
    v8f accO = cO;
#pragma unroll
    for (int kc = 0; kc < 8; ++kc) {
      accO = wmma_bf16(aFr[kc], bO[kc], accO);
    }
#pragma unroll
    for (int r = 0; r < 8; ++r) {
      int row = r + hi * 8;
      out[(((long long)(b0 + row)) * NT + t) * NO + no + col] = accO[r];
    }
    buf ^= 1;
  }
}

// ===========================================================================
// Host launcher
// ===========================================================================
extern "C" void kernel_launch(void* const* d_in, const int* in_sizes, int n_in,
                              void* d_out, int out_size, void* d_ws, size_t ws_size,
                              hipStream_t stream) {
  const float* x     = (const float*)d_in[0];
  const float* W_in  = (const float*)d_in[1];
  const float* b_in  = (const float*)d_in[2];
  const float* W_out = (const float*)d_in[3];
  const float* b_out = (const float*)d_in[4];
  const float* u_raw = (const float*)d_in[5];
  const float* sig   = (const float*)d_in[6];
  const float* v_raw = (const float*)d_in[7];
  float* out = (float*)d_out;

  char* ws = (char*)d_ws;
  float*  Umat   = (float*)(ws + WS_U);
  float*  Vmat   = (float*)(ws + WS_V);
  bf16_t* gWT    = (bf16_t*)(ws + WS_WT);
  bf16_t* gWinT  = (bf16_t*)(ws + WS_WINT);
  bf16_t* gWoutT = (bf16_t*)(ws + WS_WOUTT);

  static bool attr_done = false;
  if (!attr_done) {
    (void)hipFuncSetAttribute((const void*)build_uv_kernel,
                              hipFuncAttributeMaxDynamicSharedMemorySize,
                              L_UV_BYTES);
    (void)hipFuncSetAttribute((const void*)rnn_kernel,
                              hipFuncAttributeMaxDynamicSharedMemorySize,
                              L_RNN_BYTES);
    attr_done = true;
  }

  build_uv_kernel<<<2, 256, L_UV_BYTES, stream>>>(u_raw, v_raw, Umat, Vmat);
  combine_w_kernel<<<256, 256, 0, stream>>>(Umat, Vmat, sig, gWT);
  transpose_aux_kernel<<<256, 256, 0, stream>>>(W_in, W_out, gWinT, gWoutT);
  rnn_kernel<<<NB / RB, 256, L_RNN_BYTES, stream>>>(x, gWT, gWinT, gWoutT,
                                                    b_in, b_out, out);
}
